// ManifoldConstrainedHyperConnections_37529424232869
// MI455X (gfx1250) — matrix operand order repaired and checked by reading, not compiled, not run on hardware
//
#include <hip/hip_runtime.h>

typedef _Float16 h16;
typedef __attribute__((ext_vector_type(16))) _Float16 v16h;
typedef __attribute__((ext_vector_type(8)))  float    v8f;

union FragH { v16h v; uint4 q[2]; };

__device__ __forceinline__ float gelu_tanh(float x) {
  float x3 = x * x * x;
  return 0.5f * x * (1.0f + tanhf(0.7978845608028654f * (x + 0.044715f * x3)));
}

__device__ __forceinline__ float sigmoid_(float x) {
  return 1.0f / (1.0f + expf(-x));
}

// blockDim.x == 256 (8 waves of 32)
__device__ __forceinline__ float blockReduceSum(float v, float* red) {
  #pragma unroll
  for (int o = 16; o > 0; o >>= 1) v += __shfl_xor(v, o, 32);
  int wave = threadIdx.x >> 5, lane = threadIdx.x & 31;
  __syncthreads();
  if (lane == 0) red[wave] = v;
  __syncthreads();
  float s = 0.f;
  #pragma unroll
  for (int w = 0; w < 8; ++w) s += red[w];
  return s;
}

// ---------------------------------------------------------------------------
// Phase 1: per-token fused RMSNorm(k*d) + gate mat-vecs + sigmoid gates +
// 4x4 Sinkhorn + layer_input RMSNorm (-> f16 GEMM input) + res_out to d_out.
// One 256-thread workgroup per token; x token (k=4, d) cached in LDS.
// ---------------------------------------------------------------------------
__global__ __launch_bounds__(256)
void mhc_phase1(const float* __restrict__ x,
                const float* __restrict__ mhc_scale,
                const float* __restrict__ res_as,
                const float* __restrict__ pre_as,
                const float* __restrict__ post_as,
                const float* __restrict__ res_alpha,
                const float* __restrict__ pre_alpha,
                const float* __restrict__ post_alpha,
                const float* __restrict__ res_beta,
                const float* __restrict__ pre_beta,
                const float* __restrict__ post_beta,
                const float* __restrict__ branch_scale,
                h16*  __restrict__ Xh,
                float* __restrict__ postWS,
                float* __restrict__ out,
                int d)
{
  extern __shared__ float smem[];
  const int KD = 4 * d;
  float* xs  = smem;        // KD floats
  float* red = smem + KD;   // 256 floats scratch (wave partials)

  const int tid   = threadIdx.x;
  const int token = blockIdx.x;
  const float* xt = x + (size_t)token * KD;

  // ---- load x into LDS + sum of squares over k*d
  float ss = 0.f;
  const int perT = KD >> 8;                 // KD / 256
  for (int t = 0; t < perT; ++t) {
    int m = tid + (t << 8);
    float v = xt[m];
    xs[m] = v;
    ss += v * v;
  }
  float ssTot  = blockReduceSum(ss, red);
  float invRms = rsqrtf(ssTot / (float)KD + 1e-6f);

  // ---- fused mat-vecs: norm_x @ {pre_alpha[.,4], post_alpha[.,4], res_alpha[.,16]}
  float ap[4] = {0, 0, 0, 0};
  float ao[4] = {0, 0, 0, 0};
  float ar[16] = {0};
  for (int t = 0; t < perT; ++t) {
    int m = tid + (t << 8);
    float nx = xs[m] * invRms * mhc_scale[m];
    float4 pa = *(const float4*)(pre_alpha  + (size_t)m * 4);
    float4 qa = *(const float4*)(post_alpha + (size_t)m * 4);
    ap[0] += nx * pa.x; ap[1] += nx * pa.y; ap[2] += nx * pa.z; ap[3] += nx * pa.w;
    ao[0] += nx * qa.x; ao[1] += nx * qa.y; ao[2] += nx * qa.z; ao[3] += nx * qa.w;
    #pragma unroll
    for (int c = 0; c < 4; ++c) {
      float4 ra = *(const float4*)(res_alpha + (size_t)m * 16 + c * 4);
      ar[c * 4 + 0] += nx * ra.x;
      ar[c * 4 + 1] += nx * ra.y;
      ar[c * 4 + 2] += nx * ra.z;
      ar[c * 4 + 3] += nx * ra.w;
    }
  }

  // ---- reduce the 24 partial sums across the block
  float vals[24];
  #pragma unroll
  for (int j = 0; j < 4; ++j)  { vals[j] = ap[j]; vals[4 + j] = ao[j]; }
  #pragma unroll
  for (int j = 0; j < 16; ++j) vals[8 + j] = ar[j];
  #pragma unroll
  for (int j = 0; j < 24; ++j)
    #pragma unroll
    for (int o = 16; o > 0; o >>= 1) vals[j] += __shfl_xor(vals[j], o, 32);
  const int wave = tid >> 5, lane = tid & 31;
  __syncthreads();
  if (lane == 0) {
    #pragma unroll
    for (int j = 0; j < 24; ++j) red[wave * 24 + j] = vals[j];
  }
  __syncthreads();
  float sums[24];
  #pragma unroll
  for (int j = 0; j < 24; ++j) {
    float s = 0.f;
    #pragma unroll
    for (int w = 0; w < 8; ++w) s += red[w * 24 + j];
    sums[j] = s;
  }
  __syncthreads();

  // ---- gates
  const float pS = pre_as[0], oS = post_as[0], rS = res_as[0];
  float preG[4], postG[4];
  #pragma unroll
  for (int j = 0; j < 4; ++j) {
    preG[j]  = sigmoid_(pS * sums[j] + pre_beta[j]);
    postG[j] = 2.0f * sigmoid_(oS * sums[4 + j] + post_beta[j]);
  }
  if (tid < 4) postWS[(size_t)token * 4 + tid] = postG[tid];

  // ---- 4x4 Sinkhorn (computed redundantly per thread; trivial)
  float rm[16];
  #pragma unroll
  for (int i = 0; i < 16; ++i) rm[i] = rS * sums[8 + i] + res_beta[i];
  #pragma unroll
  for (int j = 0; j < 4; ++j) {                 // softmax over axis -2 (rows i)
    float mx = fmaxf(fmaxf(rm[j], rm[4 + j]), fmaxf(rm[8 + j], rm[12 + j]));
    float s = 0.f;
    #pragma unroll
    for (int i = 0; i < 4; ++i) { float e = expf(rm[i * 4 + j] - mx); rm[i * 4 + j] = e; s += e; }
    float inv = 1.0f / s;
    #pragma unroll
    for (int i = 0; i < 4; ++i) rm[i * 4 + j] *= inv;
  }
  for (int it = 0; it < 20; ++it) {
    #pragma unroll
    for (int i = 0; i < 4; ++i) {               // row normalize (axis -1)
      float s = rm[i * 4] + rm[i * 4 + 1] + rm[i * 4 + 2] + rm[i * 4 + 3];
      float inv = 1.0f / fmaxf(s, 1e-12f);
      rm[i * 4] *= inv; rm[i * 4 + 1] *= inv; rm[i * 4 + 2] *= inv; rm[i * 4 + 3] *= inv;
    }
    #pragma unroll
    for (int j = 0; j < 4; ++j) {               // col normalize (axis -2)
      float s = rm[j] + rm[4 + j] + rm[8 + j] + rm[12 + j];
      float inv = 1.0f / fmaxf(s, 1e-12f);
      rm[j] *= inv; rm[4 + j] *= inv; rm[8 + j] *= inv; rm[12 + j] *= inv;
    }
  }

  // ---- layer_input = sum_k x[k,:]*pre[k]; RMSNorm over d; store f16 GEMM input
  const int dper = d >> 8;                      // d / 256 (<= 16)
  float li[16];
  float ss2 = 0.f;
  for (int t = 0; t < dper; ++t) {
    int di = tid + (t << 8);
    float v = xs[di] * preG[0] + xs[d + di] * preG[1] +
              xs[2 * d + di] * preG[2] + xs[3 * d + di] * preG[3];
    li[t] = v;
    ss2 += v * v;
  }
  float ss2Tot  = blockReduceSum(ss2, red);
  float invRmsD = rsqrtf(ss2Tot / (float)d + 1e-6f);
  for (int t = 0; t < dper; ++t) {
    int di = tid + (t << 8);
    Xh[(size_t)token * d + di] = (h16)(li[t] * invRmsD * branch_scale[di]);
  }

  // ---- res_out[m,d] = sum_k x[k,d] * res_map[k,m]  -> d_out (phase 3 adds post_out)
  for (int t = 0; t < dper; ++t) {
    int di = tid + (t << 8);
    float x0 = xs[di], x1 = xs[d + di], x2 = xs[2 * d + di], x3 = xs[3 * d + di];
    float* ob = out + (size_t)token * KD + di;
    #pragma unroll
    for (int m = 0; m < 4; ++m)
      ob[(size_t)m * d] = x0 * rm[m] + x1 * rm[4 + m] + x2 * rm[8 + m] + x3 * rm[12 + m];
  }
}

// ---------------------------------------------------------------------------
// Tiled f16 WMMA GEMM: C[M,N] = A[M,K] (f16) @ W[K,N] (fp32, converted to f16
// while staging to LDS; W fits in 192MB L2 so re-reads are cheap).
// Block 128x128x64, 8 waves, each wave 32x64 = 2x4 v_wmma_f32_16x16x32_f16.
// GELU_STORE=true : Y = (h16)gelu(C)        (GEMM1)
// GELU_STORE=false: d_out[t,k,:] += C*post  (GEMM2 epilogue, on top of res_out)
// ---------------------------------------------------------------------------
#define BM 128
#define BN 128
#define BK 64

template<bool GELU_STORE>
__global__ __launch_bounds__(256)
void mhc_gemm(const h16* __restrict__ A, const float* __restrict__ W,
              int M, int N, int K,
              h16* __restrict__ Yout,
              float* __restrict__ out,
              const float* __restrict__ postWS,
              int d)
{
  __shared__ __attribute__((aligned(16))) h16 As[BM][BK + 8];
  __shared__ __attribute__((aligned(16))) h16 Bs[BN][BK + 8];   // transposed: [n][k]

  const int tid    = threadIdx.x;
  const int lane   = tid & 31;
  const int wave   = tid >> 5;
  const int wm     = (wave & 3) * 32;     // 4 wave-rows * 32 = 128
  const int wn     = (wave >> 2) * 64;    // 2 wave-cols * 64 = 128
  const int bm     = blockIdx.y * BM;
  const int bn     = blockIdx.x * BN;
  const int ln15   = lane & 15;
  const int hiHalf = (lane >> 4) & 1;     // 0: lanes 0-15, 1: lanes 16-31

  v8f acc[2][4] = {};

  for (int kt = 0; kt < K; kt += BK) {
    // stage A tile: 128x64 halves (4 x uint4 per thread)
    #pragma unroll
    for (int t = 0; t < 4; ++t) {
      int chunk = tid + (t << 8);
      int row = chunk >> 3;
      int c8  = (chunk & 7) << 3;
      uint4 v = *(const uint4*)(A + (size_t)(bm + row) * K + kt + c8);
      *(uint4*)&As[row][c8] = v;
    }
    // stage B tile transposed: W rows kt..kt+63, cols bn..bn+127 (fp32 -> f16)
    #pragma unroll
    for (int t = 0; t < 8; ++t) {
      int chunk = tid + (t << 8);
      int r  = chunk >> 5;
      int c4 = (chunk & 31) << 2;
      float4 w4 = *(const float4*)(W + (size_t)(kt + r) * N + bn + c4);
      Bs[c4 + 0][r] = (h16)w4.x;
      Bs[c4 + 1][r] = (h16)w4.y;
      Bs[c4 + 2][r] = (h16)w4.z;
      Bs[c4 + 3][r] = (h16)w4.w;
    }
    __syncthreads();

    #pragma unroll
    for (int kk = 0; kk < 2; ++kk) {      // two K=32 WMMA steps per stage
      FragH a[2], b[4];
      // A 16x32 f16 fragment: lanes 0-15 K{0-7,16-23}, lanes 16-31 K{8-15,24-31}
      #pragma unroll
      for (int mt = 0; mt < 2; ++mt) {
        int m  = wm + mt * 16 + ln15;
        int ka = kk * 32 + hiHalf * 8;
        a[mt].q[0] = *(const uint4*)&As[m][ka];
        a[mt].q[1] = *(const uint4*)&As[m][ka + 16];
      }
      // B 32x16 f16 fragment: lanes 0-15 K{0-7,16-23}, lanes 16-31 K{8-15,24-31} (Bs is [n][k])
      #pragma unroll
      for (int nt = 0; nt < 4; ++nt) {
        int n  = wn + nt * 16 + ln15;
        int kb = kk * 32 + hiHalf * 8;
        b[nt].q[0] = *(const uint4*)&Bs[n][kb];
        b[nt].q[1] = *(const uint4*)&Bs[n][kb + 16];
      }
      #pragma unroll
      for (int mt = 0; mt < 2; ++mt)
        #pragma unroll
        for (int nt = 0; nt < 4; ++nt)
          acc[mt][nt] = __builtin_amdgcn_wmma_f32_16x16x32_f16(
              false, a[mt].v, false, b[nt].v, (short)0, acc[mt][nt], false, false);
    }
    __syncthreads();
  }

  // C/D layout: VGPR r, lanes 0-15 -> M=r, lanes 16-31 -> M=r+8; N = lane&15
  if constexpr (GELU_STORE) {
    #pragma unroll
    for (int mt = 0; mt < 2; ++mt)
      #pragma unroll
      for (int r = 0; r < 8; ++r) {
        int gm = bm + wm + mt * 16 + r + hiHalf * 8;
        #pragma unroll
        for (int nt = 0; nt < 4; ++nt) {
          int gn = bn + wn + nt * 16 + ln15;
          Yout[(size_t)gm * N + gn] = (h16)gelu_tanh(acc[mt][nt][r]);
        }
      }
  } else {
    #pragma unroll
    for (int mt = 0; mt < 2; ++mt)
      #pragma unroll
      for (int r = 0; r < 8; ++r) {
        int gm = bm + wm + mt * 16 + r + hiHalf * 8;       // token
        float4 p = *(const float4*)(postWS + (size_t)gm * 4);
        #pragma unroll
        for (int nt = 0; nt < 4; ++nt) {
          int gn = bn + wn + nt * 16 + ln15;               // d column
          float v = acc[mt][nt][r];
          float* ob = out + (size_t)gm * (size_t)(4 * d) + gn;
          ob[0]              += v * p.x;
          ob[(size_t)d]      += v * p.y;
          ob[(size_t)2 * d]  += v * p.z;
          ob[(size_t)3 * d]  += v * p.w;
        }
      }
  }
}

// ---------------------------------------------------------------------------
extern "C" void kernel_launch(void* const* d_in, const int* in_sizes, int n_in,
                              void* d_out, int out_size, void* d_ws, size_t ws_size,
                              hipStream_t stream) {
  const float* x          = (const float*)d_in[0];
  const float* mhc_scale  = (const float*)d_in[1];
  const float* res_as     = (const float*)d_in[2];
  const float* pre_as     = (const float*)d_in[3];
  const float* post_as    = (const float*)d_in[4];
  const float* res_alpha  = (const float*)d_in[5];
  const float* pre_alpha  = (const float*)d_in[6];
  const float* post_alpha = (const float*)d_in[7];
  const float* res_beta   = (const float*)d_in[8];
  const float* pre_beta   = (const float*)d_in[9];
  const float* post_beta  = (const float*)d_in[10];
  const float* branch     = (const float*)d_in[11];
  const float* w1         = (const float*)d_in[12];
  const float* w2         = (const float*)d_in[13];

  const int kd  = in_sizes[1];              // k*d = 8192
  const int d   = kd / 4;                   // k = 4 (pre_alpha: kd*k)
  const int dff = in_sizes[12] / d;         // w1: d * d_ff
  const int M   = in_sizes[0] / kd;         // tokens = b*s

  float* out = (float*)d_out;

  // workspace layout: Xh[M*d] f16 | Yh[M*dff] f16 | post[M*4] f32  (~84 MB)
  h16* Xh = (h16*)d_ws;
  size_t xhB = (size_t)M * d * sizeof(h16);
  h16* Yh = (h16*)((char*)d_ws + xhB);
  size_t yhB = (size_t)M * dff * sizeof(h16);
  float* postWS = (float*)((char*)d_ws + xhB + yhB);

  size_t smem1 = ((size_t)(4 * d) + 256) * sizeof(float);
  mhc_phase1<<<dim3(M), dim3(256), smem1, stream>>>(
      x, mhc_scale, res_as, pre_as, post_as,
      res_alpha, pre_alpha, post_alpha,
      res_beta, pre_beta, post_beta, branch,
      Xh, postWS, out, d);

  // GEMM1: Yh = gelu(Xh @ w1)   [M, dff]
  mhc_gemm<true><<<dim3(dff / BN, M / BM), dim3(256), 0, stream>>>(
      Xh, w1, M, dff, d, Yh, nullptr, nullptr, d);

  // GEMM2: out[t,k,:] += (Yh @ w2) * post[t,k]   (on top of res_out)
  mhc_gemm<false><<<dim3(d / BN, M / BM), dim3(256), 0, stream>>>(
      Yh, w2, M, d, dff, nullptr, out, postWS, d);
}
